// RVQ_75935021793708
// MI455X (gfx1250) — matrix operand order repaired and checked by reading, not compiled, not run on hardware
//
#include <hip/hip_runtime.h>
#include <hip/hip_bf16.h>

typedef _Float16 h16v __attribute__((ext_vector_type(16)));
typedef _Float16 h8v  __attribute__((ext_vector_type(8)));
typedef _Float16 h4v  __attribute__((ext_vector_type(4)));
typedef _Float16 h2v  __attribute__((ext_vector_type(2)));
typedef float    f8v  __attribute__((ext_vector_type(8)));
typedef float    f4v  __attribute__((ext_vector_type(4)));

#define BTOT        524288
#define MEL         128
#define DDIM        64
#define KCB         128
#define TM          16
#define NWAVE       8
#define ITER        4                       // row-tiles per wave
#define ROWS_PER_WG (ITER * NWAVE * TM)     // 512

union H16U { h16v v; h8v h[2]; };

__device__ __forceinline__ f8v wmma32f16(h16v a, h16v b, f8v c) {
  // v_wmma_f32_16x16x32_f16: D = A(16x32 f16) * B(32x16 f16) + C(16x16 f32)
  return __builtin_amdgcn_wmma_f32_16x16x32_f16(false, a, false, b, (short)0, c,
                                                false, false);
}

// A-fragment (16x32 f16) built from a row-major *f32* LDS tile (stride floats).
// Lanes 0-15: M=lane, K = {kb+kh*8+0..7, kb+16+kh*8+0..7}, kh = lane>>4.
__device__ __forceinline__ h16v load_a_frag_f32(const float* base, int stride,
                                                int kb, int lane) {
  const int m = lane & 15, kh = lane >> 4;
  const float* p = base + m * stride + kb + kh * 8;
  f4v x0 = *(const f4v*)(p);
  f4v x1 = *(const f4v*)(p + 4);
  f4v x2 = *(const f4v*)(p + 16);
  f4v x3 = *(const f4v*)(p + 20);
  h16v r;
#pragma unroll
  for (int j = 0; j < 4; ++j) {
    r[j]      = (_Float16)x0[j];
    r[4 + j]  = (_Float16)x1[j];
    r[8 + j]  = (_Float16)x2[j];
    r[12 + j] = (_Float16)x3[j];
  }
  return r;
}

// A-fragment from row-major f16 LDS buffer (stride in halves).
__device__ __forceinline__ h16v load_a_frag(const _Float16* base, int stride,
                                            int kb, int lane) {
  const int m = lane & 15, kh = lane >> 4;
  const _Float16* p = base + m * stride + kb + kh * 8;
  H16U u;
  u.h[0] = *(const h8v*)(p);
  u.h[1] = *(const h8v*)(p + 16);
  return u.v;
}

// B-fragment (32x16 f16); source stores column n of B as contiguous row n.
__device__ __forceinline__ h16v load_b_frag(const _Float16* bt, int stride,
                                            int n0, int kb, int lane) {
  const int n = n0 + (lane & 15), kh = lane >> 4;
  const _Float16* p = bt + n * stride + kb + kh * 8;
  H16U u;
  u.h[0] = *(const h8v*)(p);
  u.h[1] = *(const h8v*)(p + 16);
  return u.v;
}

__device__ __forceinline__ void cvt_store4(_Float16* dst, f4v w) {
  h4v h;
  h[0] = (_Float16)w[0]; h[1] = (_Float16)w[1];
  h[2] = (_Float16)w[2]; h[3] = (_Float16)w[3];
  *(h4v*)dst = h;
}

// Async-DMA one 16x128 f32 tile (8 KB) global -> LDS. 16 x b128 per lane.
// INST_OFFSET applies to both LDS and global addresses (ISA 15.18.3), so one
// address pair + stepped offsets covers the tile. Tracked on ASYNCcnt.
__device__ __forceinline__ void async_tile_load(const float* gsrc, float* ldst,
                                                int lane) {
  unsigned long long g = (unsigned long long)(uintptr_t)gsrc +
                         (unsigned long long)lane * 16u;
  unsigned l = (unsigned)(uintptr_t)ldst + (unsigned)lane * 16u;
#pragma unroll
  for (int j = 0; j < 16; ++j) {
    asm volatile("global_load_async_to_lds_b128 %0, %1, off offset:%2"
                 :: "v"(l), "v"(g), "i"(j * 512) : "memory");
  }
}

__device__ __forceinline__ void wait_async0() {
  asm volatile("s_wait_asynccnt 0x0" ::: "memory");
}
__device__ __forceinline__ void wait_ds0() {
  asm volatile("s_wait_dscnt 0x0" ::: "memory");
}

__global__ void __launch_bounds__(256)
rvq_fused(const float* __restrict__ mel,
          const float* __restrict__ Win,  const float* __restrict__ bin,
          const float* __restrict__ cb0,  const float* __restrict__ cb1,
          const float* __restrict__ Wout, const float* __restrict__ bout,
          float* __restrict__ out_mel, float* __restrict__ out_idx)
{
  extern __shared__ char smem[];
  _Float16* sWin  = (_Float16*)smem;                 // 64x128 f16
  _Float16* sCb0  = sWin  + 8192;                    // 128x64 f16
  _Float16* sCb1  = sCb0  + 8192;
  _Float16* sWout = sCb1  + 8192;
  float*    sMelF = (float*)(sWout + 8192);          // 8 waves x 16x128 f32 (DMA)
  _Float16* sR    = (_Float16*)(sMelF + NWAVE * TM * MEL); // 8 x 16x64 f16
  float*    sBin  = (float*)(sR + NWAVE * TM * DDIM);
  float*    sBout = sBin  + 64;
  float*    sCn0  = sBout + 128;                     // codebook norms
  float*    sCn1  = sCn0  + 128;
  int*      sIdx  = (int*)(sCn1 + 128);              // 8 waves x 32

  const int t    = threadIdx.x;
  const int lane = t & 31;
  const int wave = t >> 5;
  const int colL = lane & 15;
  const int wgRow0 = blockIdx.x * ROWS_PER_WG;

  float*    myStF = sMelF + wave * (TM * MEL);
  _Float16* myR   = sR    + wave * (TM * DDIM);
  int*      myIdx = sIdx  + wave * 32;

  // Kick off tile-0 DMA immediately; it overlaps all of weight staging.
  async_tile_load(mel + (size_t)(wgRow0 + wave * TM) * MEL, myStF, lane);

  // ---- cooperative weight staging (f32 -> f16) + codebook norms ----
#pragma unroll
  for (int i = 0; i < 8; ++i) {
    const int f4i = i * 256 + t;                 // 2048 float4 = 8192 floats
    cvt_store4(sWin  + f4i * 4, *(const f4v*)(Win  + f4i * 4));
    cvt_store4(sCb0  + f4i * 4, *(const f4v*)(cb0  + f4i * 4));
    cvt_store4(sCb1  + f4i * 4, *(const f4v*)(cb1  + f4i * 4));
    cvt_store4(sWout + f4i * 4, *(const f4v*)(Wout + f4i * 4));
  }
  if (t < 64)  sBin[t]  = bin[t];
  if (t < 128) sBout[t] = bout[t];
  {
    const float* cb = (t < 128) ? cb0 : cb1;
    const int k = t & 127;
    float s = 0.f;
#pragma unroll
    for (int i = 0; i < 16; ++i) {
      f4v w = *(const f4v*)(cb + k * DDIM + i * 4);
      s += w[0]*w[0] + w[1]*w[1] + w[2]*w[2] + w[3]*w[3];
    }
    ((t < 128) ? sCn0 : sCn1)[k] = s;
  }
  __syncthreads();

#pragma unroll 1
  for (int it = 0; it < ITER; ++it) {
    const int r0 = wgRow0 + it * (NWAVE * TM) + wave * TM;
    const int rb = (lane >> 4) * 8;                  // this half's first row

    // ---- wait for this tile's DMA, pull A-fragments into registers ----
    wait_async0();
    h16v a1[4];
#pragma unroll
    for (int ks = 0; ks < 4; ++ks)
      a1[ks] = load_a_frag_f32(myStF, MEL, ks * 32, lane);
    // Staging buffer is dead once a1[] is materialized: drain DS, then
    // prefetch the next tile into the same buffer while we compute.
    wait_ds0();
    if (it + 1 < ITER)
      async_tile_load(mel + (size_t)(r0 + NWAVE * TM) * MEL, myStF, lane);

    // ---- z = mel @ Win^T + bin  (16x128 . 128x64) ----
    f8v zacc[4];
#pragma unroll
    for (int nt = 0; nt < 4; ++nt) {
      const float bv = sBin[nt * 16 + colL];
      f8v c;
#pragma unroll
      for (int g = 0; g < 8; ++g) c[g] = bv;
#pragma unroll
      for (int ks = 0; ks < 4; ++ks)
        c = wmma32f16(a1[ks], load_b_frag(sWin, MEL, nt * 16, ks * 32, lane), c);
      zacc[nt] = c;
    }

    // ---- residual r = z -> LDS f16 (one base, const offsets) ----
    {
      _Float16* rp = myR + rb * DDIM + colL;
#pragma unroll
      for (int nt = 0; nt < 4; ++nt)
#pragma unroll
        for (int g = 0; g < 8; ++g)
          rp[g * DDIM + nt * 16] = (_Float16)zacc[nt][g];
    }

    // ---- VQ round: argmax_k (r.c_k - 0.5||c_k||^2) == argmin dist ----
    auto vq_round = [&](const _Float16* cbh, const float* cn, int s) {
      const h16v ra0 = load_a_frag(myR, DDIM, 0,  lane);
      const h16v ra1 = load_a_frag(myR, DDIM, 32, lane);
      float bv[8]; int bi[8];
#pragma unroll
      for (int g = 0; g < 8; ++g) { bv[g] = -3.0e38f; bi[g] = 0; }
#pragma unroll
      for (int nt = 0; nt < 8; ++nt) {
        const int col = nt * 16 + colL;
        const float ci = -0.5f * cn[col];
        f8v c;
#pragma unroll
        for (int g = 0; g < 8; ++g) c[g] = ci;
        c = wmma32f16(ra0, load_b_frag(cbh, DDIM, nt * 16, 0,  lane), c);
        c = wmma32f16(ra1, load_b_frag(cbh, DDIM, nt * 16, 32, lane), c);
#pragma unroll
        for (int g = 0; g < 8; ++g)
          if (c[g] > bv[g]) { bv[g] = c[g]; bi[g] = col; }
      }
      // butterfly in each 16-lane half (rows 0-7: lanes 0-15, rows 8-15: 16-31)
#pragma unroll
      for (int m = 1; m <= 8; m <<= 1) {
#pragma unroll
        for (int g = 0; g < 8; ++g) {
          const float ov = __shfl_xor(bv[g], m, 32);
          const int   oi = __shfl_xor(bi[g], m, 32);
          if (ov > bv[g] || (ov == bv[g] && oi < bi[g])) { bv[g] = ov; bi[g] = oi; }
        }
      }
      if (colL == 0) {
        int*   ip = myIdx + s * 16 + rb;
        float* op = out_idx + (size_t)(r0 + rb) * 2 + s;   // one 64-bit base
#pragma unroll
        for (int g = 0; g < 8; ++g) {
          ip[g] = bi[g];
          op[g * 2] = (float)bi[g];                        // offset:imm stores
        }
      }
    };

    vq_round(sCb0, sCn0, 0);

    // ---- r -= cb0[idx0] (16x64, half2 pairs, cooperative in wave) ----
#pragma unroll
    for (int i = 0; i < 16; ++i) {
      const int p   = i * 32 + lane;
      const int row = p >> 5;
      const int dp  = (p & 31) * 2;
      const int idx = myIdx[row];
      h2v rv = *(const h2v*)(myR  + row * DDIM + dp);
      h2v cv = *(const h2v*)(sCb0 + idx * DDIM + dp);
      h2v o;
      o[0] = (_Float16)((float)rv[0] - (float)cv[0]);
      o[1] = (_Float16)((float)rv[1] - (float)cv[1]);
      *(h2v*)(myR + row * DDIM + dp) = o;
    }

    vq_round(sCb1, sCn1, 1);

    // ---- z_q = cb0[idx0] + cb1[idx1] (reuse residual buffer) ----
#pragma unroll
    for (int i = 0; i < 16; ++i) {
      const int p   = i * 32 + lane;
      const int row = p >> 5;
      const int dp  = (p & 31) * 2;
      const int i0 = myIdx[row], i1 = myIdx[16 + row];
      h2v c0 = *(const h2v*)(sCb0 + i0 * DDIM + dp);
      h2v c1 = *(const h2v*)(sCb1 + i1 * DDIM + dp);
      h2v o;
      o[0] = (_Float16)((float)c0[0] + (float)c1[0]);
      o[1] = (_Float16)((float)c0[1] + (float)c1[1]);
      *(h2v*)(myR + row * DDIM + dp) = o;
    }

    // ---- mel_out = z_q @ Wout^T + bout  (16x64 . 64x128) ----
    const h16v qa0 = load_a_frag(myR, DDIM, 0,  lane);
    const h16v qa1 = load_a_frag(myR, DDIM, 32, lane);
    float* ob = out_mel + (size_t)(r0 + rb) * MEL + colL;  // one 64-bit base;
                                                           // all stores use
                                                           // const offsets
#pragma unroll
    for (int nt = 0; nt < 8; ++nt) {
      const float bvv = sBout[nt * 16 + colL];
      f8v c;
#pragma unroll
      for (int g = 0; g < 8; ++g) c[g] = bvv;
      c = wmma32f16(qa0, load_b_frag(sWout, DDIM, nt * 16, 0,  lane), c);
      c = wmma32f16(qa1, load_b_frag(sWout, DDIM, nt * 16, 32, lane), c);
#pragma unroll
      for (int g = 0; g < 8; ++g)
        ob[g * MEL + nt * 16] = c[g];    // byte offset g*512 + nt*64 (imm)
    }
  }
}

extern "C" void kernel_launch(void* const* d_in, const int* in_sizes, int n_in,
                              void* d_out, int out_size, void* d_ws, size_t ws_size,
                              hipStream_t stream) {
  const float* mel  = (const float*)d_in[0];
  const float* Win  = (const float*)d_in[1];
  const float* bin  = (const float*)d_in[2];
  const float* cb0  = (const float*)d_in[3];
  const float* cb1  = (const float*)d_in[4];
  const float* Wout = (const float*)d_in[5];
  const float* bout = (const float*)d_in[6];

  float* out_mel = (float*)d_out;
  float* out_idx = out_mel + (size_t)BTOT * MEL;     // (B,2) indices as float

  const size_t shmem =
      (size_t)(8192 * 4) * sizeof(_Float16) +                 // weights f16
      (size_t)(NWAVE * TM * MEL) * sizeof(float) +            // mel DMA tiles
      (size_t)(NWAVE * TM * DDIM) * sizeof(_Float16) +        // residual
      (size_t)(64 + 128 + 128 + 128) * sizeof(float) +        // biases + norms
      (size_t)(NWAVE * 32) * sizeof(int);                     // indices
      // ~146.8 KB -> 2 workgroups per WGP (320 KB LDS)

  hipFuncSetAttribute((const void*)rvq_fused,
                      hipFuncAttributeMaxDynamicSharedMemorySize, (int)shmem);

  rvq_fused<<<BTOT / ROWS_PER_WG, 256, shmem, stream>>>(
      mel, Win, bin, cb0, cb1, Wout, bout, out_mel, out_idx);
}
